// RegulatoryNormalization_60447369724345
// MI455X (gfx1250) — matrix operand order repaired and checked by reading, not compiled, not run on hardware
//
#include <hip/hip_runtime.h>

// ---------------------------------------------------------------------------
// MI455X (gfx1250, wave32) fused transformer block.
// All matmuls via v_wmma_f32_16x16x32_bf16 (bf16 operands, f32 accum).
// ---------------------------------------------------------------------------

#define DEVINL __device__ __forceinline__

typedef __attribute__((ext_vector_type(16))) __bf16 v16bf;
typedef __attribute__((ext_vector_type(8)))  float  v8f;
typedef __attribute__((ext_vector_type(4)))  unsigned int u32x4;

union ABCast { u32x4 u[2]; v16bf v; };

DEVINL float bf2f(unsigned short h) {
    unsigned u = ((unsigned)h) << 16;
    float f; __builtin_memcpy(&f, &u, 4); return f;
}
DEVINL unsigned short f2bf(float f) {
    unsigned u; __builtin_memcpy(&u, &f, 4);
    u += 0x7FFFu + ((u >> 16) & 1u);          // round to nearest even
    return (unsigned short)(u >> 16);
}

DEVINL v8f wmma_bf16(v16bf a, v16bf b, v8f c) {
    return __builtin_amdgcn_wmma_f32_16x16x32_bf16(false, a, false, b,
                                                   (short)0, c, false, false);
}

// A fragment (16-bit 16x32, MxK): lane = row (lane&15); per-lane K set:
//   elements 0..7  -> K = kb + koff + 0..7
//   elements 8..15 -> K = kb + koff + 16..23        (koff = 8 for lanes 16..31)
DEVINL v16bf load_a16(const unsigned short* base, int lda, int m0, int kb) {
    int lane = threadIdx.x & 31;
    int m    = lane & 15;
    int koff = (lane >> 4) ? 8 : 0;
    const unsigned short* p = base + (size_t)(m0 + m) * lda + kb + koff;
    ABCast c;
    c.u[0] = *(const u32x4*)(p);
    c.u[1] = *(const u32x4*)(p + 16);
    return c.v;
}

// B fragment (16-bit 32x16, KxN) from a row-major [N][K] matrix (i.e. B = W^T):
//   lane: n = lane&15 ; e 0..7 -> K = kb + klo + e, e 8..15 -> K = kb + klo + 16 + (e-8)  (klo = 8 for lanes 16..31)
DEVINL v16bf load_b16(const unsigned short* base, int ldb, int n0, int kb) {
    int lane = threadIdx.x & 31;
    int n    = lane & 15;
    int klo  = (lane >> 4) ? 8 : 0;
    const unsigned short* p = base + (size_t)(n0 + n) * ldb + kb + klo;
    ABCast c;
    c.u[0] = *(const u32x4*)(p);
    c.u[1] = *(const u32x4*)(p + 16);
    return c.v;
}

// One wave computes a 16(M) x 64(N) tile: 4 f32 accumulators, A reused 4x per K step.
DEVINL void gemm_core(const unsigned short* A, const unsigned short* W, int K,
                      int m0, int n0, v8f acc[4]) {
    v8f z = {0.f,0.f,0.f,0.f,0.f,0.f,0.f,0.f};
    acc[0] = z; acc[1] = z; acc[2] = z; acc[3] = z;
    for (int kb = 0; kb < K; kb += 32) {
        v16bf a = load_a16(A, K, m0, kb);
#pragma unroll
        for (int t = 0; t < 4; ++t) {
            v16bf b = load_b16(W, K, n0 + 16 * t, kb);
            acc[t] = wmma_bf16(a, b, acc[t]);
        }
    }
}

// ---------------------------------------------------------------------------
// f32 -> bf16 weight conversion
// ---------------------------------------------------------------------------
__global__ __launch_bounds__(256)
void reg_cvt_kernel(const float* __restrict__ in, unsigned short* __restrict__ out, int n) {
    int i = blockIdx.x * 256 + threadIdx.x;
    if (i < n) out[i] = f2bf(in[i]);
}

// ---------------------------------------------------------------------------
// Conditional LayerNorm: one 128-thread block per token (D=512, float4 each)
// ---------------------------------------------------------------------------
__global__ __launch_bounds__(128)
void reg_ln_kernel(const float* __restrict__ x, const int* __restrict__ gt,
                   const float* __restrict__ tf_g, const float* __restrict__ tf_b,
                   const float* __restrict__ tg_g, const float* __restrict__ tg_b,
                   float* __restrict__ xn_f, unsigned short* __restrict__ xn_bf) {
    __shared__ float s1[128], s2[128];
    int tok = blockIdx.x, tid = threadIdx.x;
    const float4* xp = (const float4*)(x + (size_t)tok * 512);
    float4 v = xp[tid];
    s1[tid] = v.x + v.y + v.z + v.w;
    s2[tid] = v.x*v.x + v.y*v.y + v.z*v.z + v.w*v.w;
    __syncthreads();
    for (int o = 64; o > 0; o >>= 1) {
        if (tid < o) { s1[tid] += s1[tid + o]; s2[tid] += s2[tid + o]; }
        __syncthreads();
    }
    float mu  = s1[0] * (1.f / 512.f);
    float var = s2[0] * (1.f / 512.f) - mu * mu;
    float r   = rsqrtf(var + 1e-5f);
    int g = gt[tok];
    int base = tid * 4;
    float4 o4;
    if (g == 0 || g == 1) {
        const float* gp = (g == 0) ? tf_g : tg_g;
        const float* bp = (g == 0) ? tf_b : tg_b;
        o4.x = (v.x - mu) * r * gp[base + 0] + bp[base + 0];
        o4.y = (v.y - mu) * r * gp[base + 1] + bp[base + 1];
        o4.z = (v.z - mu) * r * gp[base + 2] + bp[base + 2];
        o4.w = (v.w - mu) * r * gp[base + 3] + bp[base + 3];
    } else {
        o4 = v;
    }
    ((float4*)(xn_f + (size_t)tok * 512))[tid] = o4;
    unsigned short* ob = xn_bf + (size_t)tok * 512 + base;
    ob[0] = f2bf(o4.x); ob[1] = f2bf(o4.y); ob[2] = f2bf(o4.z); ob[3] = f2bf(o4.w);
}

// ---------------------------------------------------------------------------
// QKV projection: C = xn @ Wqkv^T + b ; scatter Q,K as [bh][s][hd], V as [bh][hd][s]
// grid (6, 512), block 128 (4 waves x 64 cols)
// ---------------------------------------------------------------------------
__global__ __launch_bounds__(128)
void reg_qkv_kernel(const unsigned short* __restrict__ xn_bf,
                    const unsigned short* __restrict__ wqkv,
                    const float* __restrict__ bias,
                    unsigned short* __restrict__ q, unsigned short* __restrict__ k,
                    unsigned short* __restrict__ vt) {
    int wave = threadIdx.x >> 5, lane = threadIdx.x & 31;
    int n0 = blockIdx.x * 256 + wave * 64;
    int m0 = blockIdx.y * 16;
    v8f acc[4];
    gemm_core(xn_bf, wqkv, 512, m0, n0, acc);
    int off = (lane >> 4) ? 8 : 0;
    int nl  = lane & 15;
#pragma unroll
    for (int t = 0; t < 4; ++t) {
        int gn = n0 + t * 16 + nl;
        int part = gn >> 9, within = gn & 511;
        int h = within >> 6, hd = within & 63;
        float bv = bias[gn];
#pragma unroll
        for (int j = 0; j < 8; ++j) {
            int gm = m0 + j + off;
            int b = gm >> 11, s = gm & 2047;
            int bh = b * 8 + h;
            unsigned short val = f2bf(acc[t][j] + bv);
            if      (part == 0) q [((size_t)bh * 2048 + s) * 64 + hd] = val;
            else if (part == 1) k [((size_t)bh * 2048 + s) * 64 + hd] = val;
            else                vt[((size_t)bh * 64 + hd) * 2048 + s] = val;
        }
    }
}

// ---------------------------------------------------------------------------
// Flash-style attention (no max-subtraction: scaled scores are tiny; softmax is
// shift-invariant so results match the reference in fp32).
// grid (32, 8, 4) = (s-blocks, H, B); block 128 = 4 waves, 16 Q-rows per wave.
// ---------------------------------------------------------------------------
__global__ __launch_bounds__(128)
void reg_attn_kernel(const unsigned short* __restrict__ q,
                     const unsigned short* __restrict__ k,
                     const unsigned short* __restrict__ vt,
                     unsigned short* __restrict__ o_bf) {
    // per-wave 16x32 bf16 prob chunk, padded to stride 40 (80B = 20 banks, conflict-free)
    __shared__ __align__(16) unsigned short pchunk[4][16 * 40];
    int wave = threadIdx.x >> 5, lane = threadIdx.x & 31;
    int bh = blockIdx.z * 8 + blockIdx.y;
    int s0 = (blockIdx.x * 4 + wave) * 16;
    const unsigned short* qb = q  + (size_t)bh * 2048 * 64;
    const unsigned short* kb = k  + (size_t)bh * 2048 * 64;
    const unsigned short* vb = vt + (size_t)bh * 64 * 2048;
    unsigned short* pl = &pchunk[wave][0];

    // Q tile held in registers for the whole loop (K-dim = HD = 64 -> 2 fragments)
    v16bf aq0 = load_a16(qb, 64, s0, 0);
    v16bf aq1 = load_a16(qb, 64, s0, 32);

    v8f z = {0.f,0.f,0.f,0.f,0.f,0.f,0.f,0.f};
    v8f oacc[4] = {z, z, z, z};
    float rs[8] = {0.f,0.f,0.f,0.f,0.f,0.f,0.f,0.f};
    int off = (lane >> 4) ? 8 : 0;
    int nl  = lane & 15;

    for (int kc = 0; kc < 2048; kc += 32) {
        // ---- scores for 32 key columns: 2 tiles x 2 WMMAs, exp, stage to LDS ----
#pragma unroll
        for (int half = 0; half < 2; ++half) {
            int nb = kc + half * 16;
            v8f sc = z;
            sc = wmma_bf16(aq0, load_b16(kb, 64, nb, 0),  sc);
            sc = wmma_bf16(aq1, load_b16(kb, 64, nb, 32), sc);
#pragma unroll
            for (int j = 0; j < 8; ++j) {
                float e = __expf(sc[j] * 0.125f);   // /sqrt(HD)
                rs[j] += e;                          // row j+off partial sum
                pl[(j + off) * 40 + half * 16 + nl] = f2bf(e);
            }
        }
        asm volatile("s_wait_dscnt 0" ::: "memory");
        // ---- reload probs as A fragment (16x32, stride 40) ----
        const unsigned short* ap = pl + nl * 40 + off;
        ABCast ca;
        ca.u[0] = *(const u32x4*)(ap);
        ca.u[1] = *(const u32x4*)(ap + 16);
        v16bf ap16 = ca.v;
        // ---- PV: 4 output column groups of 16 (HD = 64) ----
#pragma unroll
        for (int t = 0; t < 4; ++t)
            oacc[t] = wmma_bf16(ap16, load_b16(vb, 2048, t * 16, kc), oacc[t]);
        asm volatile("s_wait_dscnt 0" ::: "memory");
    }

    // Row sums: butterfly within each 16-lane half (rows j+off match D layout)
#pragma unroll
    for (int j = 0; j < 8; ++j) {
        float v = rs[j];
        v += __shfl_xor(v, 1, 32);
        v += __shfl_xor(v, 2, 32);
        v += __shfl_xor(v, 4, 32);
        v += __shfl_xor(v, 8, 32);
        rs[j] = 1.0f / v;
    }

    int b = bh >> 3, h = bh & 7;
#pragma unroll
    for (int t = 0; t < 4; ++t)
#pragma unroll
        for (int j = 0; j < 8; ++j) {
            int s = s0 + j + off;
            int col = h * 64 + t * 16 + nl;
            o_bf[((size_t)b * 2048 + s) * 512 + col] = f2bf(oacc[t][j] * rs[j]);
        }
}

// ---------------------------------------------------------------------------
// out_proj + bias + residual -> xr (f32 and bf16). grid (2, 512), block 128.
// ---------------------------------------------------------------------------
__global__ __launch_bounds__(128)
void reg_outproj_kernel(const unsigned short* __restrict__ o_bf,
                        const unsigned short* __restrict__ wo,
                        const float* __restrict__ bias,
                        const float* __restrict__ xn_f,
                        float* __restrict__ xr_f, unsigned short* __restrict__ xr_bf) {
    int wave = threadIdx.x >> 5, lane = threadIdx.x & 31;
    int n0 = blockIdx.x * 256 + wave * 64;
    int m0 = blockIdx.y * 16;
    v8f acc[4];
    gemm_core(o_bf, wo, 512, m0, n0, acc);
    int off = (lane >> 4) ? 8 : 0;
    int nl  = lane & 15;
#pragma unroll
    for (int t = 0; t < 4; ++t) {
        int gn = n0 + t * 16 + nl;
        float bv = bias[gn];
#pragma unroll
        for (int j = 0; j < 8; ++j) {
            int gm = m0 + j + off;
            size_t idx = (size_t)gm * 512 + gn;
            float v = acc[t][j] + bv + xn_f[idx];
            xr_f[idx]  = v;
            xr_bf[idx] = f2bf(v);
        }
    }
}

// ---------------------------------------------------------------------------
// Gate layer 1: h = silu(xr @ w1^T + b1). N=256 -> grid (1, 512), block 128.
// ---------------------------------------------------------------------------
__global__ __launch_bounds__(128)
void reg_gate1_kernel(const unsigned short* __restrict__ xr_bf,
                      const unsigned short* __restrict__ w1bf,
                      const float* __restrict__ b1,
                      unsigned short* __restrict__ hbf) {
    int wave = threadIdx.x >> 5, lane = threadIdx.x & 31;
    int n0 = wave * 64;
    int m0 = blockIdx.y * 16;
    v8f acc[4];
    gemm_core(xr_bf, w1bf, 512, m0, n0, acc);
    int off = (lane >> 4) ? 8 : 0;
    int nl  = lane & 15;
#pragma unroll
    for (int t = 0; t < 4; ++t) {
        int gn = n0 + t * 16 + nl;
        float bv = b1[gn];
#pragma unroll
        for (int j = 0; j < 8; ++j) {
            int gm = m0 + j + off;
            float v = acc[t][j] + bv;
            v = v / (1.f + __expf(-v));             // SiLU
            hbf[(size_t)gm * 256 + gn] = f2bf(v);
        }
    }
}

// ---------------------------------------------------------------------------
// Gate layer 2 + final scale: one wave per token (8 waves / 256-thread block).
// ---------------------------------------------------------------------------
__global__ __launch_bounds__(256)
void reg_gate2_kernel(const unsigned short* __restrict__ hbf,
                      const float* __restrict__ w2, const float* __restrict__ b2,
                      const float* __restrict__ xr, float* __restrict__ out) {
    int wave = threadIdx.x >> 5, lane = threadIdx.x & 31;
    int tok = blockIdx.x * 8 + wave;
    const unsigned short* hp = hbf + (size_t)tok * 256 + lane * 8;
    u32x4 hu = *(const u32x4*)hp;
    const unsigned short* hs = (const unsigned short*)&hu;
    const float4* wp = (const float4*)(w2 + lane * 8);
    float4 w0 = wp[0], w1v = wp[1];
    float dot = bf2f(hs[0]) * w0.x + bf2f(hs[1]) * w0.y +
                bf2f(hs[2]) * w0.z + bf2f(hs[3]) * w0.w +
                bf2f(hs[4]) * w1v.x + bf2f(hs[5]) * w1v.y +
                bf2f(hs[6]) * w1v.z + bf2f(hs[7]) * w1v.w;
    dot += __shfl_xor(dot, 1, 32);
    dot += __shfl_xor(dot, 2, 32);
    dot += __shfl_xor(dot, 4, 32);
    dot += __shfl_xor(dot, 8, 32);
    dot += __shfl_xor(dot, 16, 32);
    float strength = 1.f / (1.f + __expf(-(dot + b2[0])));
    const float4* xp = (const float4*)(xr + (size_t)tok * 512);
    float4* op = (float4*)(out + (size_t)tok * 512);
#pragma unroll
    for (int i = 0; i < 4; ++i) {
        float4 v = xp[lane * 4 + i];
        v.x *= strength; v.y *= strength; v.z *= strength; v.w *= strength;
        op[lane * 4 + i] = v;
    }
}

// ---------------------------------------------------------------------------
extern "C" void kernel_launch(void* const* d_in, const int* in_sizes, int n_in,
                              void* d_out, int out_size, void* d_ws, size_t ws_size,
                              hipStream_t stream) {
    const float* x     = (const float*)d_in[0];
    const int*   gts   = (const int*)  d_in[1];
    const float* tf_g  = (const float*)d_in[2];
    const float* tf_b  = (const float*)d_in[3];
    const float* tg_g  = (const float*)d_in[4];
    const float* tg_b  = (const float*)d_in[5];
    const float* in_w  = (const float*)d_in[6];
    const float* in_b  = (const float*)d_in[7];
    const float* out_w = (const float*)d_in[8];
    const float* out_b = (const float*)d_in[9];
    const float* w1    = (const float*)d_in[10];
    const float* b1    = (const float*)d_in[11];
    const float* w2    = (const float*)d_in[12];
    const float* b2    = (const float*)d_in[13];

    char* ws = (char*)d_ws;
    float*          xn_f  = (float*)         (ws + 0);          // 16 MB
    float*          xr_f  = (float*)         (ws + 16777216);   // 16 MB
    unsigned short* xn_bf = (unsigned short*)(ws + 33554432);   //  8 MB
    unsigned short* qb    = (unsigned short*)(ws + 41943040);   //  8 MB
    unsigned short* kbuf  = (unsigned short*)(ws + 50331648);   //  8 MB
    unsigned short* vt    = (unsigned short*)(ws + 58720256);   //  8 MB
    unsigned short* ob    = (unsigned short*)(ws + 67108864);   //  8 MB
    unsigned short* xr_bf = (unsigned short*)(ws + 75497472);   //  8 MB
    unsigned short* hbf   = (unsigned short*)(ws + 83886080);   //  4 MB
    unsigned short* w_in  = (unsigned short*)(ws + 88080384);   // 1.5 MB
    unsigned short* w_out = (unsigned short*)(ws + 89653248);   // 0.5 MB
    unsigned short* w_1   = (unsigned short*)(ws + 90177536);   // 0.25 MB

    reg_cvt_kernel<<<(786432 + 255) / 256, 256, 0, stream>>>(in_w,  w_in,  786432);
    reg_cvt_kernel<<<(262144 + 255) / 256, 256, 0, stream>>>(out_w, w_out, 262144);
    reg_cvt_kernel<<<(131072 + 255) / 256, 256, 0, stream>>>(w1,    w_1,   131072);

    reg_ln_kernel<<<8192, 128, 0, stream>>>(x, gts, tf_g, tf_b, tg_g, tg_b, xn_f, xn_bf);

    reg_qkv_kernel<<<dim3(6, 512), 128, 0, stream>>>(xn_bf, w_in, in_b, qb, kbuf, vt);

    reg_attn_kernel<<<dim3(32, 8, 4), 128, 0, stream>>>(qb, kbuf, vt, ob);

    reg_outproj_kernel<<<dim3(2, 512), 128, 0, stream>>>(ob, w_out, out_b, xn_f, xr_f, xr_bf);

    reg_gate1_kernel<<<dim3(1, 512), 128, 0, stream>>>(xr_bf, w_1, b1, hbf);

    reg_gate2_kernel<<<1024, 256, 0, stream>>>(hbf, w2, b2, xr_f, (float*)d_out);
}
